// TopologicalValuePolicy_18769007083774
// MI455X (gfx1250) — matrix operand order repaired and checked by reading, not compiled, not run on hardware
//
#include <hip/hip_runtime.h>
#include <hip/hip_bf16.h>
#include <math.h>

typedef __attribute__((ext_vector_type(16))) _Float16     v16h;
typedef __attribute__((ext_vector_type(8)))  _Float16     v8h;
typedef __attribute__((ext_vector_type(8)))  float        v8f;
typedef __attribute__((ext_vector_type(4)))  unsigned int u32x4;
typedef __attribute__((ext_vector_type(8)))  int          i32x8;
typedef __attribute__((ext_vector_type(4)))  int          i32x4;

// Geometry
#define IMG_H   19
#define IMG_W   19
#define IMG_C   3
#define NELEM   (IMG_H * IMG_W * IMG_C)   // 1083
#define PAD_H   25                        // 17 core rows + 8 taps
#define ROW_S   80                        // padded image row stride (f16): holds 25*3=75 taps + K-pad to 32
#define K_REAL  243
#define KT_STEPS 9                        // K' = dy*32 + (dx*3+ch), 9 steps of 32
#define KST     296                       // per-channel K' stride in sBsT (f16): 288 + pad, 16B-aligned, bank-spread
#define N_ALL   48
#define N_USE   40
#define NPIX    289
#define NJOBS   57                        // 19 M-tiles * 3 N-tiles

__global__ __launch_bounds__(256)
void topo_value_policy_kernel(const float* __restrict__ state,
                              const float* __restrict__ Wf,
                              const float* __restrict__ bias,
                              float* __restrict__ probs,
                              float* __restrict__ value)
{
    __shared__ _Float16 sImg[PAD_H * ROW_S];   //  4,000 B zero-padded f16 image
    __shared__ _Float16 sBsT[N_ALL * KST];     // 28,416 B filters, N-major, K'-padded
    __shared__ float    sStage[NELEM];         //  4,332 B raw fp32 image (TDM target)
    __shared__ float    sBias[N_ALL];          //    192 B
    __shared__ _Float16 sC[NPIX * N_USE];      // 23,120 B relu(conv+b)
    __shared__ float    sLog[NPIX];            //  1,156 B
    __shared__ float    redA[256];
    __shared__ float    redB[256];

    const int tid  = threadIdx.x;
    const int lane = tid & 31;
    const int wave = tid >> 5;
    const int b    = blockIdx.x;

    const float* img = state + (size_t)b * NELEM;

    // ---- stage the raw image into LDS via the Tensor Data Mover (wave 0 issues)
#if __has_builtin(__builtin_amdgcn_tensor_load_to_lds)
    if (wave == 0) {
        unsigned long long ga = (unsigned long long)(uintptr_t)img;
        unsigned int lds_off  = (unsigned int)(uintptr_t)(void*)sStage;  // low 32b = LDS byte offset
        // D# group0: count=1 | lds_addr | global_addr[56:0] | type=2
        u32x4 g0 = { 1u, lds_off, (unsigned int)(ga & 0xFFFFFFFFu),
                     (unsigned int)(((ga >> 32) & 0x01FFFFFFu) | 0x80000000u) };
        // D# group1: data_size=4B; 1-D tensor: tensor_dim0=1083, tensor_dim1=1,
        //            tile_dim0=1083, tile_dim1=1, dim0_stride=1083
        i32x8 g1 = { 0x00020000, (int)(NELEM << 16), 0x00010000, (int)(NELEM << 16),
                     1, NELEM, 0, 0 };
        i32x4 gz = { 0, 0, 0, 0 };
#if defined(__clang_major__) && __clang_major__ >= 23
        i32x8 gz8 = { 0, 0, 0, 0, 0, 0, 0, 0 };
        __builtin_amdgcn_tensor_load_to_lds(g0, g1, gz, gz, gz8, 0);
#else
        __builtin_amdgcn_tensor_load_to_lds(g0, g1, gz, gz, 0);
#endif
    }
#else
    for (int i = tid; i < NELEM; i += 256) sStage[i] = img[i];
#endif

    // ---- overlap: zero-fill padded buffers while the TDM runs
    for (int i = tid; i < PAD_H * ROW_S; i += 256) sImg[i] = (_Float16)0.0f;
    for (int i = tid; i < N_ALL * KST;   i += 256) sBsT[i] = (_Float16)0.0f;
    if (tid < N_ALL) sBias[tid] = bias[tid];

#if __has_builtin(__builtin_amdgcn_tensor_load_to_lds)
    if (wave == 0) __builtin_amdgcn_s_wait_tensorcnt(0);
#endif
    __syncthreads();

    // ---- scatter: fp32 stage -> padded f16 image; filters -> N-major K'-padded
    for (int i = tid; i < NELEM; i += 256) {
        int y   = i / (IMG_W * IMG_C);
        int rem = i - y * (IMG_W * IMG_C);            // x*3 + ch
        sImg[(y + 3) * ROW_S + rem + 9] = (_Float16)sStage[i];
    }
    for (int i = tid; i < N_ALL * K_REAL; i += 256) {
        int n  = i / K_REAL;
        int kk = i - n * K_REAL;                      // dy*27 + dx*3 + ch
        int dy = kk / 27;
        int rm = kk - dy * 27;
        sBsT[n * KST + dy * 32 + rm] = (_Float16)Wf[kk * N_ALL + n];
    }
    __syncthreads();

    // ---- implicit-GEMM conv: (289 x 288') * (288' x 40), WMMA f16 -> f32
    const int kbA  = (lane >> 4) * 16;    // A half-wave K offset
    const int kbB  = (lane >> 4) * 16;    // B half-wave K offset
    const int ncol = lane & 15;

    for (int job = wave; job < NJOBS; job += 8) {
        int mtile = job / 3;
        int ntile = job - mtile * 3;

        int mrow = mtile * 16 + (lane & 15);
        if (mrow > NPIX - 1) mrow = NPIX - 1;         // clamped rows never stored
        int oy = mrow / 17;
        int ox = mrow - oy * 17;
        const int abase = oy * ROW_S + ox * 3 + kbA;
        const int bbase = (ntile * 16 + ncol) * KST + kbB;

        v8f acc = {};
#pragma unroll
        for (int kt = 0; kt < KT_STEPS; ++kt) {
            // A fragment: unconditional u16 gathers, all within one padded row
            v16h a;
            const int rb = abase + kt * ROW_S;        // K'-block kt == tap row dy
#pragma unroll
            for (int e = 0; e < 16; ++e) {
                int v  = e >> 1;
                int kl = 2 * v + (e & 1);                        // {0..15}
                a[e] = sImg[rb + kl];
            }
            // B fragment: 16 K'-consecutive halves = two aligned b128 loads
            const v8h* bp = (const v8h*)&sBsT[bbase + kt * 32];
            v8h b0 = bp[0];
            v8h b1 = bp[1];
            v16h bf = __builtin_shufflevector(b0, b1,
                        0, 1, 2, 3, 4, 5, 6, 7, 8, 9, 10, 11, 12, 13, 14, 15);

            acc = __builtin_amdgcn_wmma_f32_16x16x32_f16(
                    false, a, false, bf, (short)0, acc, false, false);
        }

#pragma unroll
        for (int r = 0; r < 8; ++r) {
            int M = r + ((lane >> 4) << 3);
            int P = mtile * 16 + M;
            int C = ntile * 16 + ncol;
            if (P < NPIX && C < N_USE) {
                float vv = acc[r] + sBias[C];
                sC[P * N_USE + C] = (_Float16)(vv > 0.0f ? vv : 0.0f);
            }
        }
    }
    __syncthreads();

    // ---- epilogue: (5,2,4) power sums -> logits & value; softmax
    float localMax = -3.4e38f;
    float localV   = 0.0f;
    for (int p = tid; p < NPIX; p += 256) {
        const _Float16* c = &sC[p * N_USE];
        float fg[2];
#pragma unroll
        for (int g = 0; g < 2; ++g) {
            float sd = 0.0f;
#pragma unroll
            for (int d = 0; d < 4; ++d) {
                float s = 0.0f;
#pragma unroll
                for (int q = 0; q < 5; ++q) {
                    float x  = (float)c[q * 8 + g * 4 + d];
                    float x2 = x * x;
                    s += x2 * x2 * x2;                        // x^6
                }
                sd += (s > 0.0f) ? __expf((5.0f / 6.0f) * __logf(s)) : 0.0f;
            }
            fg[g] = (sd > 0.0f) ? __expf(0.2f * __logf(sd)) : 0.0f;
        }
        float lg = fg[0] + fg[1];
        sLog[p]  = lg;
        localV  += fg[0] - fg[1];
        localMax = fmaxf(localMax, lg);
    }
    redA[tid] = localMax;
    redB[tid] = localV;
    __syncthreads();
    for (int s = 128; s > 0; s >>= 1) {
        if (tid < s) {
            redA[tid] = fmaxf(redA[tid], redA[tid + s]);
            redB[tid] += redB[tid + s];
        }
        __syncthreads();
    }
    const float m    = redA[0];
    const float vsum = redB[0];
    __syncthreads();

    float localE = 0.0f;
    for (int p = tid; p < NPIX; p += 256) {
        float e = __expf(2.0f * (sLog[p] - m));      // POLICY_STRETCH = 2
        sLog[p] = e;
        localE += e;
    }
    redA[tid] = localE;
    __syncthreads();
    for (int s = 128; s > 0; s >>= 1) {
        if (tid < s) redA[tid] += redA[tid + s];
        __syncthreads();
    }
    const float inv = 1.0f / redA[0];

    float* po = probs + (size_t)b * NPIX;
    for (int p = tid; p < NPIX; p += 256) po[p] = sLog[p] * inv;
    if (tid == 0) value[b] = tanhf((0.2f / 32.0f) * vsum);   // GAUGE * STRETCH
}

extern "C" void kernel_launch(void* const* d_in, const int* in_sizes, int n_in,
                              void* d_out, int out_size, void* d_ws, size_t ws_size,
                              hipStream_t stream) {
    (void)n_in; (void)out_size; (void)d_ws; (void)ws_size;
    const float* state = (const float*)d_in[0];   // (B,19,19,3) fp32
    const float* Wf    = (const float*)d_in[1];   // (9,9,3,48)  fp32
    const float* bias  = (const float*)d_in[2];   // (48,)       fp32

    const int B = in_sizes[0] / NELEM;            // 4096

    float* probs = (float*)d_out;                 // (B,289)
    float* value = probs + (size_t)B * NPIX;      // (B,)

    topo_value_policy_kernel<<<B, 256, 0, stream>>>(state, Wf, bias, probs, value);
}
